// MultiHeadAttentionLayer_73418170958472
// MI455X (gfx1250) — compile-verified, matrix operand order checked by reading, not run on hardware
//
#include <hip/hip_runtime.h>
#include <hip/hip_bf16.h>
#include <math.h>

// ---------------------------------------------------------------------------
// MI455X / gfx1250 implementation. bf16 WMMA (16x16x32, f32 accum) for all
// GEMMs; attention fully fused per 16-query tile (scores never leave LDS).
// K-tiles in attention are staged into LDS by the Tensor Data Mover
// (tensor_load_to_lds + s_wait_tensorcnt) with per-wave double buffering.
// ---------------------------------------------------------------------------

typedef __bf16 bf16;
typedef __attribute__((ext_vector_type(16))) __bf16 v16bf;
typedef __attribute__((ext_vector_type(8)))  __bf16 bf16x8;
typedef __attribute__((ext_vector_type(8)))  float  v8f;
typedef unsigned int u32x4 __attribute__((ext_vector_type(4)));
typedef int          i32x4 __attribute__((ext_vector_type(4)));
typedef int          i32x8 __attribute__((ext_vector_type(8)));

#define HID   128
#define NHEAD 8
#define FFND  512
#define BATCH 32
#define SEQ   512
#define NTOK  (BATCH * SEQ)              // 16384 tokens
#define ATTN_SCALE 0.08838834764831845f  // 128^-0.5

// Dynamic-LDS layout for the attention kernel (byte offsets from 0 so the
// TDM descriptor lds_addr fields are known constants).
#define SM_S    0                         // float S[16][512]   : 32768 B
#define SM_P    32768                     // bf16  P[16][512]   : 16384 B
#define SM_RMAX 49152                     // float [16][8]      :   512 B
#define SM_RSUM 49664                     // float [16][8]      :   512 B
#define SM_KST  50176                     // bf16  kstage[4][2][16*128] : 32768 B
#define ATTN_SMEM (SM_KST + 4 * 2 * 16 * HID * 2)   // 82944 B

#if __has_builtin(__builtin_amdgcn_tensor_load_to_lds)
#define HAS_TDM 1
#else
#define HAS_TDM 0
#endif

// ---------------------------------------------------------------------------
// TDM: async 2D tile load (16 rows x 128 bf16, row stride 128) into LDS.
// D# built per CDNA5 ISA (group0/group1 bit layout).
// ---------------------------------------------------------------------------
__device__ __forceinline__ void tdm_load_k_tile(unsigned lds_off, const bf16* gptr) {
#if HAS_TDM
  unsigned long long ga = (unsigned long long)(uintptr_t)gptr;
  u32x4 g0 = { 1u,                                    // count=1 (valid D#)
               lds_off,                                // lds_addr (bytes)
               (unsigned)ga,                           // global_addr[31:0]
               (unsigned)(ga >> 32) | (2u << 30) };    // addr[56:32] | type=2
  i32x8 g1 = { 1 << 16,            // workgroup_mask=0 | data_size=1 (2B)
               128 << 16,          // tensor_dim0[15:0]=128 in bits [63:48]
               512 << 16,          // tensor_dim1[15:0]=512 in bits [95:80]
               128 << 16,          // tile_dim0=128 in bits [127:112]
               16,                 // tile_dim1=16, tile_dim2=0
               128,                // tensor_dim0_stride = 128 elements
               0, 0 };             // tensor_dim1_stride (unused, 2D)
  i32x4 g2 = { 0, 0, 0, 0 };
  i32x4 g3 = { 0, 0, 0, 0 };
#if __clang_major__ >= 23
  i32x8 g4 = { 0, 0, 0, 0, 0, 0, 0, 0 };
  __builtin_amdgcn_tensor_load_to_lds(g0, g1, g2, g3, g4, 0);
#else
  __builtin_amdgcn_tensor_load_to_lds(g0, g1, g2, g3, 0);
#endif
#else
  (void)lds_off; (void)gptr;
#endif
}

// s_wait_tensorcnt needs an immediate -> template constant.
template <int N>
__device__ __forceinline__ void tdm_wait() {
#if __has_builtin(__builtin_amdgcn_s_wait_tensorcnt)
  __builtin_amdgcn_s_wait_tensorcnt(N);
#endif
}

// ---------------------------------------------------------------------------
// Fragment loader for 16-bit A/B operands of v_wmma_f32_16x16x32_bf16.
// Row-major [16 x ld] tile; lanes 0-15 row (lane&15) K-groups {0..7,16..23},
// lanes 16-31 same rows, K-groups {8..15,24..31}: two b128 loads per lane.
// ---------------------------------------------------------------------------
__device__ __forceinline__ v16bf load_frag(const bf16* base, int ld) {
  int lane = threadIdx.x & 31;
  int half = lane >> 4;
  const bf16* p = base + (size_t)(lane & 15) * ld + half * 8;
  bf16x8 lo = *(const bf16x8*)(p);
  bf16x8 hi = *(const bf16x8*)(p + 16);
  v16bf f;
#pragma unroll
  for (int j = 0; j < 8; ++j) { f[j] = lo[j]; f[j + 8] = hi[j]; }
  return f;
}

__device__ __forceinline__ v8f wmma_bf16(v16bf a, v16bf b, v8f c) {
  return __builtin_amdgcn_wmma_f32_16x16x32_bf16(false, a, false, b, (short)0, c,
                                                 false, false);
}

__device__ __forceinline__ v8f zero8() {
  v8f z = {0.f, 0.f, 0.f, 0.f, 0.f, 0.f, 0.f, 0.f};
  return z;
}

// ---------------------------------------------------------------------------
// Weight pre-pack: f32 [K,N] row-major  ->  bf16 [N,K] row-major (B^T)
// ---------------------------------------------------------------------------
__global__ void transpose_to_bf16(const float* __restrict__ in,
                                  bf16* __restrict__ out, int K, int N) {
  int idx = blockIdx.x * blockDim.x + threadIdx.x;
  if (idx >= K * N) return;
  int k = idx / N, n = idx - k * N;
  out[(size_t)n * K + k] = (bf16)in[idx];
}

// ---------------------------------------------------------------------------
// LayerNorm over HID=128, one wave32 per token (4 feats/lane), bf16 output.
// ---------------------------------------------------------------------------
__global__ __launch_bounds__(128) void ln_kernel(const float* __restrict__ x,
                                                 const float* __restrict__ g,
                                                 const float* __restrict__ bb,
                                                 bf16* __restrict__ out) {
  int tok  = blockIdx.x * 4 + (threadIdx.x >> 5);
  int lane = threadIdx.x & 31;
  const float* row = x + (size_t)tok * HID;
  float v[4];
#pragma unroll
  for (int i = 0; i < 4; ++i) v[i] = row[lane + 32 * i];
  float s = v[0] + v[1] + v[2] + v[3];
#pragma unroll
  for (int o = 16; o > 0; o >>= 1) s += __shfl_xor(s, o, 32);
  float mean = s * (1.0f / 128.0f);
  float q = 0.f;
#pragma unroll
  for (int i = 0; i < 4; ++i) { float d = v[i] - mean; q += d * d; }
#pragma unroll
  for (int o = 16; o > 0; o >>= 1) q += __shfl_xor(q, o, 32);
  float rstd = rsqrtf(q * (1.0f / 128.0f) + 1e-5f);
#pragma unroll
  for (int i = 0; i < 4; ++i) {
    int c = lane + 32 * i;
    out[(size_t)tok * HID + c] = (bf16)((v[i] - mean) * rstd * g[c] + bb[c]);
  }
}

// ---------------------------------------------------------------------------
// Fused QKV projection. One 16x32 output tile per wave (2 accumulators,
// shared A fragment), K=128 (4 k-steps). Pair index spans 96:
// [0,32)=Q [32,64)=K [64,96)=V. Applies mask (and SCALE for Q).
// V is stored transposed (feature-major).
// ---------------------------------------------------------------------------
__global__ __launch_bounds__(128) void qkv_kernel(
    const bf16* __restrict__ xln, const bf16* __restrict__ WqT,
    const bf16* __restrict__ WkT, const bf16* __restrict__ WvT,
    const float* __restrict__ bq, const float* __restrict__ bk,
    const float* __restrict__ bv, const float* __restrict__ mask,
    bf16* __restrict__ Q, bf16* __restrict__ Kx, bf16* __restrict__ VT) {
  int wid = blockIdx.x * 4 + (threadIdx.x >> 5);
  int mt = wid / 96;
  int pp = wid - mt * 96;
  int which = pp >> 5;                 // 0=Q 1=K 2=V (wave-uniform)
  int nloc = (pp & 31) * 32;           // feature base of the 32-wide pair
  const bf16* WT   = (which == 0) ? WqT : (which == 1) ? WkT : WvT;
  const float* bia = (which == 0) ? bq  : (which == 1) ? bk  : bv;

  const bf16* arow  = xln + (size_t)(mt * 16) * HID;
  const bf16* brow0 = WT + (size_t)nloc * HID;
  const bf16* brow1 = brow0 + (size_t)16 * HID;
  v8f acc0 = zero8(), acc1 = zero8();
#pragma unroll
  for (int kk = 0; kk < 4; ++kk) {
    v16bf a = load_frag(arow + kk * 32, HID);
    acc0 = wmma_bf16(a, load_frag(brow0 + kk * 32, HID), acc0);
    acc1 = wmma_bf16(a, load_frag(brow1 + kk * 32, HID), acc1);
  }

  int lane = threadIdx.x & 31;
  int half = lane >> 4;
#pragma unroll
  for (int s = 0; s < 2; ++s) {
    int n    = nloc + s * 16 + (lane & 15);
    int h    = n >> 7;
    int dloc = n & 127;
    float bn = bia[n];
    const v8f& acc = s ? acc1 : acc0;
#pragma unroll
    for (int r = 0; r < 8; ++r) {
      int tok  = mt * 16 + 8 * half + r;
      int b    = tok >> 9;
      int srow = tok & 511;
      float mv  = mask[tok];
      float val = acc[r] + bn;
      size_t hb = ((size_t)b * NHEAD + h);
      if (which == 0)      Q [(hb * SEQ + srow) * HID + dloc] = (bf16)(val * mv * ATTN_SCALE);
      else if (which == 1) Kx[(hb * SEQ + srow) * HID + dloc] = (bf16)(val * mv);
      else                 VT[(hb * HID + dloc) * SEQ + srow] = (bf16)(val * mv);
    }
  }
}

// ---------------------------------------------------------------------------
// Fused attention: one block (4 waves) per (b, h, 16-query tile).
//   S = q k^T + dist, mask -> softmax in LDS -> P bf16 -> y = P V.
// K tiles staged via TDM double buffering (per-wave), overlapped with WMMA.
// ---------------------------------------------------------------------------
__global__ __launch_bounds__(128) void attn_kernel(
    const bf16* __restrict__ Q, const bf16* __restrict__ Kx,
    const bf16* __restrict__ VT, const float* __restrict__ dist,
    const float* __restrict__ mask, bf16* __restrict__ Y) {
  extern __shared__ char smem[];
  float (*S)[SEQ]    = (float(*)[SEQ])(smem + SM_S);
  bf16  (*P)[SEQ]    = (bf16 (*)[SEQ])(smem + SM_P);
  float (*redmax)[8] = (float(*)[8])(smem + SM_RMAX);
  float (*redsum)[8] = (float(*)[8])(smem + SM_RSUM);
  bf16* kstage       = (bf16*)(smem + SM_KST);

  int bh   = blockIdx.x >> 5;          // b*NHEAD + h
  int qt   = blockIdx.x & 31;
  int b    = bh >> 3;
  int h    = bh & 7;
  int wave = threadIdx.x >> 5;
  int lane = threadIdx.x & 31;
  int half = lane >> 4;
  const float* maskb = mask + b * SEQ;

  // --- Phase 1: S tiles (each wave owns 8 key tiles) -----------------------
  const bf16* qbase = Q + ((size_t)bh * SEQ + qt * 16) * HID;
  v16bf qf[4];
#pragma unroll
  for (int kk = 0; kk < 4; ++kk) qf[kk] = load_frag(qbase + kk * 32, HID);

  const bf16* khead = Kx + (size_t)bh * SEQ * HID;

  auto do_tile = [&](int kt, const bf16* ktile /* [16 x 128] row-major */) {
    v8f acc = zero8();
#pragma unroll
    for (int kk = 0; kk < 4; ++kk)
      acc = wmma_bf16(qf[kk], load_frag(ktile + kk * 32, HID), acc);
    int n = kt * 16 + (lane & 15);
    float mk = maskb[n];
#pragma unroll
    for (int r = 0; r < 8; ++r) {
      int m    = 8 * half + r;
      int qrow = qt * 16 + m;
      float sv = acc[r] + dist[((size_t)b * SEQ + qrow) * SEQ + n];
      if (maskb[qrow] * mk == 0.f) sv = -1.0e9f;
      S[m][n] = sv;
    }
  };

  if (HAS_TDM) {
    // per-wave double-buffered TDM staging of 16x128 bf16 K tiles
    unsigned kst_off0 = SM_KST + (unsigned)(wave * 2) * (16 * HID * 2);
    unsigned kst_off1 = kst_off0 + 16 * HID * 2;
    tdm_load_k_tile(kst_off0, khead + (size_t)(wave * 8) * 16 * HID);
    for (int t = 0; t < 8; ++t) {
      int kt = wave * 8 + t;
      if (t < 7) {
        tdm_load_k_tile((t & 1) ? kst_off0 : kst_off1,
                        khead + (size_t)(kt + 1) * 16 * HID);
        tdm_wait<1>();       // oldest (tile t) complete, next still in flight
      } else {
        tdm_wait<0>();       // last tile complete
      }
      const bf16* kb = kstage + (size_t)(wave * 2 + (t & 1)) * (16 * HID);
      do_tile(kt, kb);
    }
  } else {
#pragma unroll 2
    for (int t = 0; t < 8; ++t) {
      int kt = wave * 8 + t;
      do_tile(kt, khead + (size_t)kt * 16 * HID);
    }
  }
  __syncthreads();

  // --- Phase 2: row softmax (8 threads per row) ----------------------------
  {
    int row = threadIdx.x >> 3, sub = threadIdx.x & 7;
    float mx = -3.4e38f;
    for (int n = sub; n < SEQ; n += 8) mx = fmaxf(mx, S[row][n]);
    redmax[row][sub] = mx;
    __syncthreads();
    float rmax = redmax[row][0];
#pragma unroll
    for (int j = 1; j < 8; ++j) rmax = fmaxf(rmax, redmax[row][j]);
    float sum = 0.f;
    for (int n = sub; n < SEQ; n += 8) {
      float e = __expf(S[row][n] - rmax);
      S[row][n] = e;
      sum += e;
    }
    redsum[row][sub] = sum;
    __syncthreads();
    float rsum = 0.f;
#pragma unroll
    for (int j = 0; j < 8; ++j) rsum += redsum[row][j];
    float inv = 1.f / rsum;
    float mq  = maskb[qt * 16 + row];
    for (int n = sub; n < SEQ; n += 8) {
      float p = S[row][n] * inv;
      if (mq * maskb[n] == 0.f) p = 0.f;
      P[row][n] = (bf16)p;
    }
  }
  __syncthreads();

  // --- Phase 3: y = P V, two feature tiles per wave sharing the P fragment -
  const bf16* pl  = &P[0][0];
  const bf16* vb0 = VT + ((size_t)bh * HID + (wave * 2) * 16) * SEQ;
  const bf16* vb1 = vb0 + (size_t)16 * SEQ;
  v8f acc0 = zero8(), acc1 = zero8();
#pragma unroll 4
  for (int kk = 0; kk < 16; ++kk) {
    v16bf pa = load_frag(pl + kk * 32, SEQ);
    acc0 = wmma_bf16(pa, load_frag(vb0 + kk * 32, SEQ), acc0);
    acc1 = wmma_bf16(pa, load_frag(vb1 + kk * 32, SEQ), acc1);
  }
#pragma unroll
  for (int s = 0; s < 2; ++s) {
    int n = (wave * 2 + s) * 16 + (lane & 15);
    const v8f& acc = s ? acc1 : acc0;
#pragma unroll
    for (int r = 0; r < 8; ++r) {
      int qrow = qt * 16 + 8 * half + r;
      Y[((size_t)b * SEQ + qrow) * (NHEAD * HID) + h * HID + n] = (bf16)acc[r];
    }
  }
}

// ---------------------------------------------------------------------------
// Output projection + residual:  ATT = gcn + (Y @ Wo + bo) * mask
// One 16x32 tile per wave, K=1024 (32 k-steps, shared A fragment).
// ---------------------------------------------------------------------------
__global__ __launch_bounds__(128) void proj_kernel(
    const bf16* __restrict__ Y, const bf16* __restrict__ WoT,
    const float* __restrict__ bo, const float* __restrict__ mask,
    const float* __restrict__ gcn, float* __restrict__ ATT) {
  int wid = blockIdx.x * 4 + (threadIdx.x >> 5);
  int mt = wid >> 2, pp = wid & 3;
  int nloc = pp * 32;
  const bf16* arow  = Y + (size_t)(mt * 16) * (NHEAD * HID);
  const bf16* brow0 = WoT + (size_t)nloc * (NHEAD * HID);
  const bf16* brow1 = brow0 + (size_t)16 * (NHEAD * HID);
  v8f acc0 = zero8(), acc1 = zero8();
#pragma unroll 4
  for (int kk = 0; kk < 32; ++kk) {
    if (kk + 8 < 32) __builtin_prefetch(arow + (kk + 8) * 32, 0, 0);
    v16bf a = load_frag(arow + kk * 32, NHEAD * HID);
    acc0 = wmma_bf16(a, load_frag(brow0 + kk * 32, NHEAD * HID), acc0);
    acc1 = wmma_bf16(a, load_frag(brow1 + kk * 32, NHEAD * HID), acc1);
  }
  int lane = threadIdx.x & 31;
  int half = lane >> 4;
#pragma unroll
  for (int s = 0; s < 2; ++s) {
    int n = nloc + s * 16 + (lane & 15);
    float bn = bo[n];
    const v8f& acc = s ? acc1 : acc0;
#pragma unroll
    for (int r = 0; r < 8; ++r) {
      int tok = mt * 16 + 8 * half + r;
      size_t idx = (size_t)tok * HID + n;
      ATT[idx] = gcn[idx] + (acc[r] + bn) * mask[tok];
    }
  }
}

// ---------------------------------------------------------------------------
// FFN layer 1 + exact GELU:  H1 = gelu(F @ W1 + b1)   (bf16 out)
// 16x32 tile per wave, K=128.
// ---------------------------------------------------------------------------
__global__ __launch_bounds__(128) void ffn1_kernel(
    const bf16* __restrict__ F, const bf16* __restrict__ W1T,
    const float* __restrict__ b1, bf16* __restrict__ H1) {
  int wid = blockIdx.x * 4 + (threadIdx.x >> 5);
  int mt = wid >> 4, pp = wid & 15;
  int nloc = pp * 32;
  const bf16* arow  = F + (size_t)(mt * 16) * HID;
  const bf16* brow0 = W1T + (size_t)nloc * HID;
  const bf16* brow1 = brow0 + (size_t)16 * HID;
  v8f acc0 = zero8(), acc1 = zero8();
#pragma unroll
  for (int kk = 0; kk < 4; ++kk) {
    v16bf a = load_frag(arow + kk * 32, HID);
    acc0 = wmma_bf16(a, load_frag(brow0 + kk * 32, HID), acc0);
    acc1 = wmma_bf16(a, load_frag(brow1 + kk * 32, HID), acc1);
  }
  int lane = threadIdx.x & 31;
  int half = lane >> 4;
#pragma unroll
  for (int s = 0; s < 2; ++s) {
    int n = nloc + s * 16 + (lane & 15);
    float bn = b1[n];
    const v8f& acc = s ? acc1 : acc0;
#pragma unroll
    for (int r = 0; r < 8; ++r) {
      float x = acc[r] + bn;
      float gl = 0.5f * x * (1.f + erff(x * 0.70710678118654752f));
      H1[(size_t)(mt * 16 + 8 * half + r) * FFND + n] = (bf16)gl;
    }
  }
}

// ---------------------------------------------------------------------------
// FFN layer 2 + final residuals:  out = gcn + ATT + (H1 @ W2 + b2) * mask
// 16x32 tile per wave, K=512.
// ---------------------------------------------------------------------------
__global__ __launch_bounds__(128) void ffn2_kernel(
    const bf16* __restrict__ H1, const bf16* __restrict__ W2T,
    const float* __restrict__ b2, const float* __restrict__ mask,
    const float* __restrict__ gcn, const float* __restrict__ ATT,
    float* __restrict__ out) {
  int wid = blockIdx.x * 4 + (threadIdx.x >> 5);
  int mt = wid >> 2, pp = wid & 3;
  int nloc = pp * 32;
  const bf16* arow  = H1 + (size_t)(mt * 16) * FFND;
  const bf16* brow0 = W2T + (size_t)nloc * FFND;
  const bf16* brow1 = brow0 + (size_t)16 * FFND;
  v8f acc0 = zero8(), acc1 = zero8();
#pragma unroll 4
  for (int kk = 0; kk < 16; ++kk) {
    v16bf a = load_frag(arow + kk * 32, FFND);
    acc0 = wmma_bf16(a, load_frag(brow0 + kk * 32, FFND), acc0);
    acc1 = wmma_bf16(a, load_frag(brow1 + kk * 32, FFND), acc1);
  }
  int lane = threadIdx.x & 31;
  int half = lane >> 4;
#pragma unroll
  for (int s = 0; s < 2; ++s) {
    int n = nloc + s * 16 + (lane & 15);
    float bn = b2[n];
    const v8f& acc = s ? acc1 : acc0;
#pragma unroll
    for (int r = 0; r < 8; ++r) {
      int tok = mt * 16 + 8 * half + r;
      size_t idx = (size_t)tok * HID + n;
      out[idx] = gcn[idx] + ATT[idx] + (acc[r] + bn) * mask[tok];
    }
  }
}

// ---------------------------------------------------------------------------
// Launcher
// ---------------------------------------------------------------------------
extern "C" void kernel_launch(void* const* d_in, const int* in_sizes, int n_in,
                              void* d_out, int out_size, void* d_ws, size_t ws_size,
                              hipStream_t stream) {
  (void)in_sizes; (void)n_in; (void)out_size; (void)ws_size;
  const float* gcn   = (const float*)d_in[0];
  const float* mask  = (const float*)d_in[1];
  const float* dist  = (const float*)d_in[2];
  const float* ln1_g = (const float*)d_in[3];
  const float* ln1_b = (const float*)d_in[4];
  const float* Wq    = (const float*)d_in[5];
  const float* bq    = (const float*)d_in[6];
  const float* Wk    = (const float*)d_in[7];
  const float* bk    = (const float*)d_in[8];
  const float* Wv    = (const float*)d_in[9];
  const float* bv    = (const float*)d_in[10];
  const float* Wo    = (const float*)d_in[11];
  const float* bo    = (const float*)d_in[12];
  const float* ln2_g = (const float*)d_in[13];
  const float* ln2_b = (const float*)d_in[14];
  const float* W1    = (const float*)d_in[15];
  const float* b1    = (const float*)d_in[16];
  const float* W2    = (const float*)d_in[17];
  const float* b2    = (const float*)d_in[18];

  // Workspace carve-up (256B aligned slabs), ~170 MB total.
  size_t off = 0;
  auto carve = [&](size_t bytes) -> void* {
    void* p = (char*)d_ws + off;
    off += (bytes + 255) & ~(size_t)255;
    return p;
  };
  bf16* WqT = (bf16*)carve((size_t)HID * NHEAD * HID * 2);
  bf16* WkT = (bf16*)carve((size_t)HID * NHEAD * HID * 2);
  bf16* WvT = (bf16*)carve((size_t)HID * NHEAD * HID * 2);
  bf16* WoT = (bf16*)carve((size_t)NHEAD * HID * HID * 2);
  bf16* W1T = (bf16*)carve((size_t)HID * FFND * 2);
  bf16* W2T = (bf16*)carve((size_t)FFND * HID * 2);
  bf16* XLN = (bf16*)carve((size_t)NTOK * HID * 2);
  bf16* Qb  = (bf16*)carve((size_t)BATCH * NHEAD * SEQ * HID * 2);
  bf16* Kb  = (bf16*)carve((size_t)BATCH * NHEAD * SEQ * HID * 2);
  bf16* VTb = (bf16*)carve((size_t)BATCH * NHEAD * HID * SEQ * 2);
  bf16* Yb  = (bf16*)carve((size_t)NTOK * NHEAD * HID * 2);
  float* ATTb = (float*)carve((size_t)NTOK * HID * 4);
  bf16* Fb  = (bf16*)carve((size_t)NTOK * HID * 2);
  bf16* H1b = (bf16*)carve((size_t)NTOK * FFND * 2);

  // 1) weight pre-pack (bf16, transposed)
  transpose_to_bf16<<<(HID * NHEAD * HID + 255) / 256, 256, 0, stream>>>(Wq, WqT, HID, NHEAD * HID);
  transpose_to_bf16<<<(HID * NHEAD * HID + 255) / 256, 256, 0, stream>>>(Wk, WkT, HID, NHEAD * HID);
  transpose_to_bf16<<<(HID * NHEAD * HID + 255) / 256, 256, 0, stream>>>(Wv, WvT, HID, NHEAD * HID);
  transpose_to_bf16<<<(NHEAD * HID * HID + 255) / 256, 256, 0, stream>>>(Wo, WoT, NHEAD * HID, HID);
  transpose_to_bf16<<<(HID * FFND + 255) / 256, 256, 0, stream>>>(W1, W1T, HID, FFND);
  transpose_to_bf16<<<(FFND * HID + 255) / 256, 256, 0, stream>>>(W2, W2T, FFND, HID);

  // 2) LN1
  ln_kernel<<<NTOK / 4, 128, 0, stream>>>(gcn, ln1_g, ln1_b, XLN);

  // 3) QKV: (NTOK/16) * 96 tile-pairs / 4 waves per block
  qkv_kernel<<<(NTOK / 16) * 96 / 4, 128, 0, stream>>>(XLN, WqT, WkT, WvT, bq, bk, bv,
                                                       mask, Qb, Kb, VTb);

  // 4) fused attention: one block per (b, h, q-tile); dynamic LDS (TDM stage)
  attn_kernel<<<BATCH * NHEAD * (SEQ / 16), 128, ATTN_SMEM, stream>>>(Qb, Kb, VTb, dist,
                                                                      mask, Yb);

  // 5) output projection + residual
  proj_kernel<<<(NTOK / 16) * 4 / 4, 128, 0, stream>>>(Yb, WoT, bo, mask, gcn, ATTb);

  // 6) LN2
  ln_kernel<<<NTOK / 4, 128, 0, stream>>>(ATTb, ln2_g, ln2_b, Fb);

  // 7) FFN
  ffn1_kernel<<<(NTOK / 16) * 16 / 4, 128, 0, stream>>>(Fb, W1T, b1, H1b);
  ffn2_kernel<<<(NTOK / 16) * 4 / 4, 128, 0, stream>>>(H1b, W2T, b2, mask, gcn,
                                                       ATTb, (float*)d_out);
}
